// SAModule_34849364640185
// MI455X (gfx1250) — matrix operand order repaired and checked by reading, not compiled, not run on hardware
//
#include <hip/hip_runtime.h>
#include <hip/hip_bf16.h>

typedef __attribute__((ext_vector_type(16))) _Float16 v16h;
typedef __attribute__((ext_vector_type(8)))  _Float16 v8h;
typedef __attribute__((ext_vector_type(8)))  float    v8f;

// ---------------- problem constants ----------------
#define BG    8
#define NPTS  2048
#define FIN   64
#define KNB   32          // neighbors per node == wave32
#define MSMP  512         // fps samples per graph
#define H1D   128
#define H2D   256
#define GOUTD 256
#define NNODE (BG*NPTS)   // 16384
#define R2    (0.1f*0.1f)

// padded K dims for WMMA (multiples of 32)
#define K1PAD 96          // 67 -> 96
#define K2PAD 128
#define KGPAD 256

// ---------------- workspace layout (bytes) ----------------
#define FPS_OFF   ((size_t)0)                       // BG*MSMP int32           = 16384
#define KNN_OFF   ((size_t)16384)                   // NNODE*KNB int32         = 2097152
#define VALID_OFF (KNN_OFF + (size_t)2097152)       // NNODE u32               = 65536
#define W1P_OFF   (VALID_OFF + (size_t)65536)       // 3*8*512 f16             = 24576
#define W2P_OFF   (W1P_OFF + (size_t)24576)         // 4*16*512 f16            = 65536
#define WGP_OFF   (W2P_OFF + (size_t)65536)         // 8*16*512 f16            = 131072
#define AGG_OFF   (WGP_OFF + (size_t)131072)        // NNODE*H2D f32           = 16777216

// ---------------- d_out layout (elements) ----------------
#define XALL_SZ   ((size_t)NNODE * GOUTD)           // 4194304
#define XDST_OFF  (XALL_SZ)                         // 4096*256
#define POS_OFF   (XDST_OFF + (size_t)BG*MSMP*GOUTD)
#define BATCH_OFF (POS_OFF + (size_t)BG*MSMP*3)

// ---------------- WMMA helpers ----------------
__device__ inline v8f wmma16(v16h a, v16h b, v8f c) {
    return __builtin_amdgcn_wmma_f32_16x16x32_f16(
        /*neg_a=*/false, a, /*neg_b=*/false, b,
        /*c_mod=*/(short)0, c, /*reuse_a=*/false, /*reuse_b=*/false);
}

// A fragment from LDS row-major f16 buffer [rows][stride] per ISA 16-bit A layout:
// lanes 0-15: row = rowbase+lane, K chunks {k..k+7, k+16..k+23}
// lanes 16-31: row = rowbase+(lane-16), K chunks {k+8..k+15, k+24..k+31}
__device__ inline v16h load_a_lds(const _Float16* buf, int stride, int rowbase,
                                  int lane, int kbase) {
    int row = rowbase + (lane & 15);
    int k0  = kbase + ((lane & 16) ? 8 : 0);
    const _Float16* p = buf + row * stride + k0;
    v8h lo = *(const v8h*)(p);
    v8h hi = *(const v8h*)(p + 16);
    v16h a;
#pragma unroll
    for (int e = 0; e < 8; ++e) { a[e] = lo[e]; a[e + 8] = hi[e]; }
    return a;
}

// B fragment: packed lane-major, 512 halves per fragment, 16 contiguous halves/lane
__device__ inline v16h load_b_frag(const _Float16* packed, int frag, int lane) {
    return *(const v16h*)(packed + (size_t)frag * 512 + (size_t)lane * 16);
}

// ---------------- kernel 1: pack weights to f16 B-fragments ----------------
// B layout: lane<16 -> col = nt*16+lane, K = kt*32 + {0..7, 16..23}
//           lane>=16 -> col = nt*16+lane-16, K = kt*32 + {8..15, 24..31}
__global__ __launch_bounds__(256) void pack_weights(const float* __restrict__ W,
                                                    _Float16* __restrict__ out,
                                                    int Kreal, int Kpad, int N) {
    int tid = blockIdx.x * 256 + threadIdx.x;
    int NT = N >> 4, KT = Kpad >> 5;
    int total = KT * NT * 512;
    if (tid >= total) return;
    int e    = tid & 15;
    int lane = (tid >> 4) & 31;
    int frag = tid >> 9;
    int nt = frag % NT, kt = frag / NT;
    int n = nt * 16 + (lane & 15);
    int k = kt * 32 + ((lane & 16) ? 8 : 0) + e + ((e & 8) ? 8 : 0);
    float v = (k < Kreal) ? W[(size_t)k * N + n] : 0.0f;
    out[tid] = (_Float16)v;
}

// ---------------- kernel 2: farthest point sampling ----------------
__global__ __launch_bounds__(256) void fps_kernel(const float* __restrict__ pos,
                                                  int* __restrict__ fps_idx) {
    __shared__ float px[NPTS], py[NPTS], pz[NPTS];
    __shared__ float rv[256];
    __shared__ int   ri[256];
    __shared__ int   s_last;
    int g = blockIdx.x, t = threadIdx.x;
    for (int j = t; j < NPTS; j += 256) {
        const float* p = pos + (size_t)(g * NPTS + j) * 3;
        px[j] = p[0]; py[j] = p[1]; pz[j] = p[2];
    }
    __syncthreads();
    float mind[8];
#pragma unroll
    for (int u = 0; u < 8; ++u) mind[u] = 1e30f;
    int last = 0;
    if (t == 0) fps_idx[g * MSMP] = 0;
    for (int m = 1; m < MSMP; ++m) {
        float lx = px[last], ly = py[last], lz = pz[last];
        float best = -1.0f; int bi = 0;
#pragma unroll
        for (int u = 0; u < 8; ++u) {
            int j = t + u * 256;
            float dx = px[j] - lx, dy = py[j] - ly, dz = pz[j] - lz;
            float d = dx * dx + dy * dy + dz * dz;
            mind[u] = fminf(mind[u], d);
            if (mind[u] > best) { best = mind[u]; bi = j; }
        }
        rv[t] = best; ri[t] = bi;
        __syncthreads();
        for (int s = 128; s > 0; s >>= 1) {
            if (t < s) {
                float ov = rv[t + s]; int oi = ri[t + s];
                if (ov > rv[t] || (ov == rv[t] && oi < ri[t])) { rv[t] = ov; ri[t] = oi; }
            }
            __syncthreads();
        }
        if (t == 0) { s_last = ri[0]; fps_idx[g * MSMP + m] = ri[0]; }
        __syncthreads();
        last = s_last;
    }
}

// ---------------- kernel 3: KNN-within-radius (one wave per node) ----------------
#define KW 4
__global__ __launch_bounds__(128) void knn_kernel(const float* __restrict__ pos,
                                                  int* __restrict__ knn,
                                                  unsigned* __restrict__ validm) {
    __shared__ float px[NPTS], py[NPTS], pz[NPTS];
    __shared__ float d2[KW][NPTS];
    int blocksPerGraph = NPTS / KW;                  // 512
    int g = blockIdx.x / blocksPerGraph;
    int nodebase = (blockIdx.x % blocksPerGraph) * KW;
    int wave = threadIdx.x >> 5, lane = threadIdx.x & 31;
    for (int j = threadIdx.x; j < NPTS; j += 128) {
        const float* p = pos + (size_t)(g * NPTS + j) * 3;
        px[j] = p[0]; py[j] = p[1]; pz[j] = p[2];
    }
    __syncthreads();
    int i = nodebase + wave;
    float ix = px[i], iy = py[i], iz = pz[i];
    for (int j = lane; j < NPTS; j += 32) {
        float dx = px[j] - ix, dy = py[j] - iy, dz = pz[j] - iz;
        float d = dx * dx + dy * dy + dz * dz;
        if (j == i) d = 1e9f;
        d2[wave][j] = d;
    }
    __syncthreads();
    unsigned mask = 0;
    int myidx = 0;
    for (int kk = 0; kk < KNB; ++kk) {
        float best = 1e30f; int bi = 0;
#pragma unroll
        for (int u = 0; u < NPTS / 32; ++u) {
            int j = lane + u * 32;
            float d = d2[wave][j];
            if (d < best) { best = d; bi = j; }
        }
#pragma unroll
        for (int s = 16; s > 0; s >>= 1) {
            float ov = __shfl_xor(best, s, 32);
            int   oi = __shfl_xor(bi, s, 32);
            if (ov < best || (ov == best && oi < bi)) { best = ov; bi = oi; }
        }
        if (lane == kk) myidx = bi;
        if (best <= R2) mask |= (1u << kk);
        if (lane == (bi & 31)) d2[wave][bi] = 1e30f;   // invalidate winner
    }
    knn[(size_t)(g * NPTS + i) * KNB + lane] = myidx;
    if (lane == 0) validm[g * NPTS + i] = mask;
}

// ---------------- kernel 4: edge MLP + masked max aggregation (WMMA) ----------------
#define EW 4
__global__ __launch_bounds__(128) void edge_mlp_kernel(
    const float* __restrict__ x, const float* __restrict__ pos,
    const int* __restrict__ knn, const unsigned* __restrict__ validm,
    const _Float16* __restrict__ W1p, const _Float16* __restrict__ W2p,
    const float* __restrict__ b1, const float* __restrict__ b2,
    float* __restrict__ agg) {
    __shared__ _Float16 feat[EW][KNB][K1PAD];   // 24 KB
    __shared__ _Float16 h1s[EW][KNB][H1D];      // 32 KB
    int wave = threadIdx.x >> 5, lane = threadIdx.x & 31;
    int node = blockIdx.x * EW + wave;
    int g = node >> 11;
    unsigned mask = validm[node];
    int j = knn[(size_t)node * KNB + lane];

    __builtin_prefetch(W2p, 0, 1);              // pull weights toward L2/L1

    // build edge-feature rows (lane == neighbor row)
    const float* xj = x + (size_t)(g * NPTS + j) * FIN;
#pragma unroll
    for (int c = 0; c < FIN; ++c) feat[wave][lane][c] = (_Float16)xj[c];
    const float* pj = pos + (size_t)(g * NPTS + j) * 3;
    const float* pi = pos + (size_t)node * 3;
    feat[wave][lane][FIN + 0] = (_Float16)(pj[0] - pi[0]);
    feat[wave][lane][FIN + 1] = (_Float16)(pj[1] - pi[1]);
    feat[wave][lane][FIN + 2] = (_Float16)(pj[2] - pi[2]);
#pragma unroll
    for (int c = FIN + 3; c < K1PAD; ++c) feat[wave][lane][c] = (_Float16)0.0f;

    int colbase = lane & 15;
    int halfrow = (lane >> 4) << 3;             // 0 or 8

    // ---- layer 1: [32 x 96] @ [96 x 128] ----
    for (int mt = 0; mt < 2; ++mt) {
        v8f acc[8];
#pragma unroll
        for (int q = 0; q < 8; ++q) acc[q] = (v8f){0,0,0,0,0,0,0,0};
#pragma unroll
        for (int kt = 0; kt < 3; ++kt) {
            v16h a = load_a_lds(&feat[wave][0][0], K1PAD, mt * 16, lane, kt * 32);
#pragma unroll
            for (int nt = 0; nt < 8; ++nt) {
                v16h b = load_b_frag(W1p, kt * 8 + nt, lane);
                acc[nt] = wmma16(a, b, acc[nt]);
            }
        }
        int rowoff = mt * 16 + halfrow;
#pragma unroll
        for (int nt = 0; nt < 8; ++nt) {
            int col = nt * 16 + colbase;
            float bb = b1[col];
#pragma unroll
            for (int r = 0; r < 8; ++r) {
                float h = acc[nt][r] + bb;
                h = h > 0.0f ? h : 0.0f;
                h1s[wave][rowoff + r][col] = (_Float16)h;
            }
        }
    }

    // ---- layer 2: [32 x 128] @ [128 x 256], masked max over 32 rows ----
    float colmax[16];
#pragma unroll
    for (int q = 0; q < 16; ++q) colmax[q] = 0.0f;   // relu>=0, 0-init == PyG max aggr
    for (int mt = 0; mt < 2; ++mt) {
        for (int ntg = 0; ntg < 2; ++ntg) {
            v8f acc[8];
#pragma unroll
            for (int q = 0; q < 8; ++q) acc[q] = (v8f){0,0,0,0,0,0,0,0};
#pragma unroll
            for (int kt = 0; kt < 4; ++kt) {
                v16h a = load_a_lds(&h1s[wave][0][0], H1D, mt * 16, lane, kt * 32);
#pragma unroll
                for (int q = 0; q < 8; ++q) {
                    int nt = ntg * 8 + q;
                    v16h b = load_b_frag(W2p, kt * 16 + nt, lane);
                    acc[q] = wmma16(a, b, acc[q]);
                }
            }
            int rowoff = mt * 16 + halfrow;
#pragma unroll
            for (int q = 0; q < 8; ++q) {
                int nt = ntg * 8 + q;
                int col = nt * 16 + colbase;
                float bb = b2[col];
                float m = 0.0f;
#pragma unroll
                for (int r = 0; r < 8; ++r) {
                    int row = rowoff + r;                  // neighbor index
                    float h = acc[q][r] + bb;
                    h = h > 0.0f ? h : 0.0f;
                    if ((mask >> row) & 1u) m = fmaxf(m, h);
                }
                m = fmaxf(m, __shfl_xor(m, 16, 32));       // merge row halves (same col)
                colmax[nt] = fmaxf(colmax[nt], m);
            }
        }
    }
    if (lane < 16) {
#pragma unroll
        for (int nt = 0; nt < 16; ++nt)
            agg[(size_t)node * H2D + nt * 16 + lane] = colmax[nt];
    }
}

// ---------------- kernel 5: global linear  out = agg @ Wg + bg (WMMA) ----------------
__global__ __launch_bounds__(256) void global_linear_kernel(
    const float* __restrict__ agg, const _Float16* __restrict__ Wgp,
    const float* __restrict__ bg, float* __restrict__ xall) {
    int wave = threadIdx.x >> 5, lane = threadIdx.x & 31;
    int mt = blockIdx.x * 8 + wave;            // 1024 row tiles
    int rowbase = mt * 16;
    int colbase = lane & 15;
    int halfrow = (lane >> 4) << 3;
    for (int ntg = 0; ntg < 4; ++ntg) {
        v8f acc[4];
#pragma unroll
        for (int q = 0; q < 4; ++q) acc[q] = (v8f){0,0,0,0,0,0,0,0};
        for (int kt = 0; kt < 8; ++kt) {
            // A fragment gathered straight from f32 agg (convert to f16)
            int row = rowbase + (lane & 15);
            int k0  = kt * 32 + ((lane & 16) ? 8 : 0);
            const float* p = agg + (size_t)row * H2D + k0;
            v16h a;
#pragma unroll
            for (int e = 0; e < 8; ++e) {
                a[e]     = (_Float16)p[e];
                a[e + 8] = (_Float16)p[e + 16];
            }
#pragma unroll
            for (int q = 0; q < 4; ++q) {
                int nt = ntg * 4 + q;
                v16h b = load_b_frag(Wgp, kt * 16 + nt, lane);
                acc[q] = wmma16(a, b, acc[q]);
            }
        }
        int rowoff = rowbase + halfrow;
#pragma unroll
        for (int q = 0; q < 4; ++q) {
            int nt = ntg * 4 + q;
            int col = nt * 16 + colbase;
            float bb = bg[col];
#pragma unroll
            for (int r = 0; r < 8; ++r)
                xall[(size_t)(rowoff + r) * GOUTD + col] = acc[q][r] + bb;
        }
    }
}

// ---------------- kernel 6: gather sampled outputs ----------------
__global__ __launch_bounds__(256) void gather_kernel(const float* __restrict__ pos,
                                                     const int* __restrict__ fps_idx,
                                                     float* __restrict__ out) {
    int s = blockIdx.x;                 // 0..4095
    int t = threadIdx.x;
    int g = s >> 9;                     // 512 samples per graph
    int li = fps_idx[s];
    int gi = g * NPTS + li;
    out[XDST_OFF + (size_t)s * GOUTD + t] = out[(size_t)gi * GOUTD + t];
    if (t < 3) out[POS_OFF + (size_t)s * 3 + t] = pos[(size_t)gi * 3 + t];
    if (t == 0) ((int*)out)[BATCH_OFF + s] = g;
}

// ---------------- launch ----------------
extern "C" void kernel_launch(void* const* d_in, const int* in_sizes, int n_in,
                              void* d_out, int out_size, void* d_ws, size_t ws_size,
                              hipStream_t stream) {
    const float* x   = (const float*)d_in[0];
    const float* pos = (const float*)d_in[1];
    // d_in[2] = batch (implicit from layout, unused)
    const float* W1 = (const float*)d_in[3];
    const float* b1 = (const float*)d_in[4];
    const float* W2 = (const float*)d_in[5];
    const float* b2 = (const float*)d_in[6];
    const float* Wg = (const float*)d_in[7];
    const float* bg = (const float*)d_in[8];

    char* ws = (char*)d_ws;
    int*      fps_idx = (int*)(ws + FPS_OFF);
    int*      knn     = (int*)(ws + KNN_OFF);
    unsigned* validm  = (unsigned*)(ws + VALID_OFF);
    _Float16* W1p     = (_Float16*)(ws + W1P_OFF);
    _Float16* W2p     = (_Float16*)(ws + W2P_OFF);
    _Float16* Wgp     = (_Float16*)(ws + WGP_OFF);
    float*    agg     = (float*)(ws + AGG_OFF);
    float*    out     = (float*)d_out;

    pack_weights<<<48, 256, 0, stream>>>(W1, W1p, FIN + 3, K1PAD, H1D);
    pack_weights<<<128, 256, 0, stream>>>(W2, W2p, H1D, K2PAD, H2D);
    pack_weights<<<256, 256, 0, stream>>>(Wg, Wgp, H2D, KGPAD, GOUTD);

    fps_kernel<<<BG, 256, 0, stream>>>(pos, fps_idx);
    knn_kernel<<<BG * (NPTS / KW), 128, 0, stream>>>(pos, knn, validm);
    edge_mlp_kernel<<<NNODE / EW, 128, 0, stream>>>(x, pos, knn, validm,
                                                    W1p, W2p, b1, b2, agg);
    global_linear_kernel<<<NNODE / (16 * 8), 256, 0, stream>>>(agg, Wgp, bg, out);
    gather_kernel<<<BG * MSMP, 256, 0, stream>>>(pos, fps_idx, out);
}
